// MultiHeadAttn_66288525247107
// MI455X (gfx1250) — matrix operand order repaired and checked by reading, not compiled, not run on hardware
//
#include <hip/hip_runtime.h>

// Flash-attention, local-window causal variant for MI455X (gfx1250, wave32).
// One wave owns a 16-query tile; 4 independent waves/block (no barriers,
// wave-synchronous LDS). K/V tiles are double-buffered in LDS and filled by
// CDNA5 async global->LDS copies (ASYNCcnt pipeline) so HBM latency overlaps
// the v_wmma_f32_16x16x32_f16 math. Memory floor: ~67MB / 23.3TB/s ~= 3us;
// windowed compute is only ~4.3 GFLOP, so this is a pure data-movement play.
//
// NOTE: LDS buffer selection is done with INTEGER offsets into the single
// extern __shared__ array (not pointer selects) so clang's addrspace
// inference keeps every access on the DS pipe (ds_load_*, s_wait_dscnt)
// instead of degrading to FLAT (loadcnt+dscnt).

typedef _Float16 h16;
typedef __attribute__((ext_vector_type(16))) _Float16 v16h;
typedef __attribute__((ext_vector_type(8)))  float    v8f;

#define HEADS 16
#define SEQ   4096
#define DIM   64
#define WIN   256
#define WAVES 4

#define TILE_F32   (32 * DIM)               // one 32x64 f32 tile = 2048 floats
#define WAVE_F32   (4 * TILE_F32)           // K[2] + V[2] double buffers
#define SMEM_BYTES (WAVES * WAVE_F32 * 4)   // 131072 B (< 320KB WGP LDS)

// One CDNA5 async copy: every lane moves 16B global->LDS, no VGPR data.
__device__ __forceinline__ void async_copy_b128(unsigned ldsOff,
                                                const float* gptr) {
  asm volatile("global_load_async_to_lds_b128 %0, %1, off"
               :
               : "v"(ldsOff), "v"((unsigned long long)(uintptr_t)gptr)
               : "memory");
}

// Stage one 32x64 f32 tile (8KB) = 16 async b128 instructions.
__device__ __forceinline__ void stage_tile_async(unsigned ldsBase,
                                                 const float* gsrc, int lane) {
  #pragma unroll
  for (int it = 0; it < 16; ++it) {
    const int idx  = it * 32 + lane;   // 512 float4 per tile
    const int row  = idx >> 4;         // 16 float4 per 64-wide row
    const int col4 = (idx & 15) * 4;
    async_copy_b128(ldsBase + (unsigned)(row * DIM + col4) * 4u,
                    gsrc + (size_t)row * DIM + col4);
  }
}

__global__ __launch_bounds__(32 * WAVES) void fa_local_kernel(
    const float* __restrict__ Q, const float* __restrict__ K,
    const float* __restrict__ V, const int* __restrict__ layer_idx,
    float* __restrict__ O)
{
  extern __shared__ float lds[];                 // K/V double buffers
  __shared__ h16 PsArr[WAVES * 16 * 32];         // P restripe tiles (static)

  const int tid  = threadIdx.x;
  const int wave = tid >> 5;
  const int lane = tid & 31;
  const int half = lane >> 4;   // 0: lanes 0-15, 1: lanes 16-31
  const int ln   = lane & 15;   // M (A/C row) or N (B/C col) within half

  // integer float-offsets into lds[]: K bufs at +0,+TILE; V at +2T,+3T
  const int wbase = wave * WAVE_F32;
  h16* Ps = PsArr + wave * (16 * 32);

  // LDS byte offsets for async VDST (low 32 bits of the shared address).
  const unsigned ldsByteBase = (unsigned)(uintptr_t)(void*)&lds[0];
  unsigned kOffs[2], vOffs[2];
  kOffs[0] = ldsByteBase + (unsigned)(wbase + 0 * TILE_F32) * 4u;
  kOffs[1] = ldsByteBase + (unsigned)(wbase + 1 * TILE_F32) * 4u;
  vOffs[0] = ldsByteBase + (unsigned)(wbase + 2 * TILE_F32) * 4u;
  vOffs[1] = ldsByteBase + (unsigned)(wbase + 3 * TILE_F32) * 4u;

  const int h  = blockIdx.y;
  const int q0 = (blockIdx.x * WAVES + wave) * 16;
  const bool windowed = (layer_idx[0] & 1) != 0;

  const size_t headOff = (size_t)h * SEQ * DIM;
  const float* Qh = Q + headOff;
  const float* Kh = K + headOff;
  const float* Vh = V + headOff;
  float*       Oh = O + headOff;

  // ---- Q fragments in A-matrix layout with contiguous k order (16x32 f16), 2 chunks over D=64 --
  // lane<16: M=ln, VGPR p holds K = 2p; lanes>=16 add +16.
  v16h aq[2];
  #pragma unroll
  for (int dc = 0; dc < 2; ++dc) {
    #pragma unroll
    for (int p = 0; p < 8; ++p) {
      const int k0 = 2*p + half*16;
      const float* src = Qh + (size_t)(q0 + ln) * DIM + dc*32 + k0;
      aq[dc][2*p+0] = (h16)src[0];
      aq[dc][2*p+1] = (h16)src[1];
    }
  }

  // ---- online-softmax state: 8 rows per lane (C-layout row = r + 8*half) --
  float m_run[8], l_run[8];
  v8f o_acc[4] = {};   // O tile 16x64 as 4 f32 accumulators
  #pragma unroll
  for (int r = 0; r < 8; ++r) { m_run[r] = -1e30f; l_run[r] = 0.f; }

  const int i_hi = q0 + 15;
  int jmin = windowed ? (q0 - (WIN - 1)) : 0;
  if (jmin < 0) jmin = 0;
  const int kt0 = jmin & ~31;

  // ---- prologue: async-stage first K/V tile into buffer 0 ----
  stage_tile_async(kOffs[0], Kh + (size_t)kt0 * DIM, lane);
  stage_tile_async(vOffs[0], Vh + (size_t)kt0 * DIM, lane);

  for (int kt = kt0; kt <= i_hi; kt += 32) {
    const int  cur      = ((kt - kt0) >> 5) & 1;
    const bool has_next = (kt + 32) <= i_hi;
    const int  kb       = wbase + cur * TILE_F32;            // K buf (floats)
    const int  vb       = wbase + (2 + cur) * TILE_F32;      // V buf (floats)

    // WAR guard: all ds reads of the buffer we are about to overwrite must
    // have retired before new async writes can land there.
    asm volatile("s_wait_dscnt 0x0" ::: "memory");
    if (has_next) {
      stage_tile_async(kOffs[cur ^ 1], Kh + (size_t)(kt + 32) * DIM, lane);
      stage_tile_async(vOffs[cur ^ 1], Vh + (size_t)(kt + 32) * DIM, lane);
      // 32 instrs now in flight for tile t+1; in-order completion means
      // "<=32 outstanding" proves tile t has fully landed in LDS.
      asm volatile("s_wait_asynccnt 0x20" ::: "memory");
    } else {
      asm volatile("s_wait_asynccnt 0x0" ::: "memory");
    }

    // ---- scores: S = Q*K^T, two 16x16 key tiles, K-dim 64 = 2 chunks ----
    // B layout (32x16 f16): lane N=ln, VGPR p holds K = 2p + 16*half.
    // f32->f16 conversion here (ds_load_b64 + v_cvt_pk_f16_f32).
    v8f s_lo = {}, s_hi = {};
    #pragma unroll
    for (int dc = 0; dc < 2; ++dc) {
      v16h b_lo, b_hi;
      #pragma unroll
      for (int p = 0; p < 8; ++p) {
        const int kd = dc*32 + 2*p + half*16;
        const int lo = kb + (0  + ln)*DIM + kd;
        const int hi = kb + (16 + ln)*DIM + kd;
        b_lo[2*p+0]=(h16)lds[lo+0]; b_lo[2*p+1]=(h16)lds[lo+1];
        b_hi[2*p+0]=(h16)lds[hi+0]; b_hi[2*p+1]=(h16)lds[hi+1];
      }
      s_lo = __builtin_amdgcn_wmma_f32_16x16x32_f16(false, aq[dc], false, b_lo,
                                                    (short)0, s_lo, false, false);
      s_hi = __builtin_amdgcn_wmma_f32_16x16x32_f16(false, aq[dc], false, b_hi,
                                                    (short)0, s_hi, false, false);
    }

    // ---- mask + scale (1/sqrt(64)=0.125), tile row-max ----
    float p_lo[8], p_hi[8], tmax[8], tsum[8];
    #pragma unroll
    for (int r = 0; r < 8; ++r) {
      const int i  = q0 + r + 8*half;
      const int j0 = kt + ln;
      const int j1 = kt + 16 + ln;
      const bool v0 = (j0 <= i) && (!windowed || (j0 > i - WIN));
      const bool v1 = (j1 <= i) && (!windowed || (j1 > i - WIN));
      p_lo[r] = v0 ? s_lo[r] * 0.125f : -1e30f;
      p_hi[r] = v1 ? s_hi[r] * 0.125f : -1e30f;
      tmax[r] = fmaxf(p_lo[r], p_hi[r]);
    }
    #pragma unroll
    for (int m = 1; m < 16; m <<= 1) {
      #pragma unroll
      for (int r = 0; r < 8; ++r)
        tmax[r] = fmaxf(tmax[r], __shfl_xor(tmax[r], m, 32));
    }

    // ---- rescale running state, exponentiate ----
    #pragma unroll
    for (int r = 0; r < 8; ++r) {
      const float m_new = fmaxf(m_run[r], tmax[r]);
      const float alpha = __expf(m_run[r] - m_new);
      p_lo[r] = (p_lo[r] > -1e29f) ? __expf(p_lo[r] - m_new) : 0.f;
      p_hi[r] = (p_hi[r] > -1e29f) ? __expf(p_hi[r] - m_new) : 0.f;
      tsum[r] = p_lo[r] + p_hi[r];
      l_run[r] *= alpha;
      m_run[r] = m_new;
      #pragma unroll
      for (int n = 0; n < 4; ++n) o_acc[n][r] *= alpha;
    }
    #pragma unroll
    for (int m = 1; m < 16; m <<= 1) {
      #pragma unroll
      for (int r = 0; r < 8; ++r)
        tsum[r] += __shfl_xor(tsum[r], m, 32);
    }
    #pragma unroll
    for (int r = 0; r < 8; ++r) l_run[r] += tsum[r];

    // ---- P: C-layout -> LDS -> A-layout restripe ----
    #pragma unroll
    for (int r = 0; r < 8; ++r) {
      Ps[(r + 8*half)*32 + ln]      = (h16)p_lo[r];
      Ps[(r + 8*half)*32 + 16 + ln] = (h16)p_hi[r];
    }
    asm volatile("s_wait_dscnt 0x0" ::: "memory");
    v16h ap;
    #pragma unroll
    for (int p = 0; p < 8; ++p) {
      const int k0 = 2*p + half*16;
      const h16* src = Ps + ln*32 + k0;
      ap[2*p+0] = src[0];
      ap[2*p+1] = src[1];
    }

    // ---- O += P * V (4 d-chunks of 16 cols) ----
    #pragma unroll
    for (int n = 0; n < 4; ++n) {
      v16h bv;
      #pragma unroll
      for (int p = 0; p < 8; ++p) {
        const int krel = 2*p + half*16;
        const int src = vb + krel*DIM + n*16 + ln;
        bv[2*p+0] = (h16)lds[src];
        bv[2*p+1] = (h16)lds[src + DIM];
      }
      o_acc[n] = __builtin_amdgcn_wmma_f32_16x16x32_f16(false, ap, false, bv,
                                                        (short)0, o_acc[n], false, false);
    }
  }

  // ---- finalize: O / l, store (16 consecutive floats per half-wave) ----
  #pragma unroll
  for (int r = 0; r < 8; ++r) {
    const float inv = (l_run[r] > 0.f) ? (1.f / l_run[r]) : 0.f;
    #pragma unroll
    for (int n = 0; n < 4; ++n) {
      Oh[(size_t)(q0 + r + 8*half)*DIM + n*16 + ln] = o_acc[n][r] * inv;
    }
  }
}

extern "C" void kernel_launch(void* const* d_in, const int* in_sizes, int n_in,
                              void* d_out, int out_size, void* d_ws, size_t ws_size,
                              hipStream_t stream) {
  const float* q = (const float*)d_in[0];
  const float* k = (const float*)d_in[1];
  const float* v = (const float*)d_in[2];
  const int* li  = (const int*)d_in[3];
  float* out     = (float*)d_out;

  dim3 grid(SEQ / (16 * WAVES), HEADS, 1);
  fa_local_kernel<<<grid, 32 * WAVES, SMEM_BYTES, stream>>>(q, k, v, li, out);
}
